// MemoryBank_41772851921156
// MI455X (gfx1250) — matrix operand order repaired and checked by reading, not compiled, not run on hardware
//
#include <hip/hip_runtime.h>
#include <math.h>

// ---------------------------------------------------------------------------
// MemoryBank.read() fused kernel for gfx1250 (MI455X), bf16 WMMA path.
// Shapes (fixed by reference): B=32, S=4096 -> ROWS=131072, D=256, N=64, K=8
// Each wave owns TWO 16-row M-tiles (32 rows) so every B fragment loaded from
// the L2-hot weight/key/memory matrices feeds two WMMAs (halves L2 traffic,
// doubles independent accumulator chains per load).
// ---------------------------------------------------------------------------

typedef __attribute__((ext_vector_type(16))) __bf16 v16bf;
typedef __attribute__((ext_vector_type(8)))  float  v8f;

union bfu {
    v16bf v;
    unsigned short s[16];
    unsigned int   u[8];
};

__device__ __forceinline__ unsigned short f2bf(float f) {
    unsigned int u = __float_as_uint(f);
    unsigned int r = u + 0x7FFFu + ((u >> 16) & 1u);   // round-to-nearest-even
    return (unsigned short)(r >> 16);
}

#define WMMA_BF16(a, b, c) \
    __builtin_amdgcn_wmma_f32_16x16x32_bf16(false, (a), false, (b), (short)0, (c), false, false)

// ---------------------------------------------------------------------------
// Prep kernel (1 workgroup): convert W_q to bf16, compute k = mem @ W_k^T
// (bf16, row-major [n][d] so K-pairs along d are dword-contiguous), memT bf16
// ([d][n] so K-pairs along n are dword-contiguous), and the importance bias.
// ---------------------------------------------------------------------------
__global__ __launch_bounds__(256)
void mb_prep(const float* __restrict__ mem, const float* __restrict__ imp,
             const float* __restrict__ age, const float* __restrict__ Wq,
             const float* __restrict__ Wk,
             unsigned short* __restrict__ wq_bf,     // [256][256]
             unsigned short* __restrict__ k_bf,      // [64][256]
             unsigned short* __restrict__ memT_bf,   // [256][64]
             float* __restrict__ bias)               // [64]
{
    const int t = threadIdx.x;

    // W_q -> bf16, natural [d_out][k] layout (K-pairs contiguous)
    for (int i = t; i < 256 * 256; i += 256) wq_bf[i] = f2bf(Wq[i]);

    // k[n][d_out] = sum_i mem[n][i] * W_k[d_out][i]   (tiny: 4.2 MFLOP)
    for (int e = t; e < 64 * 256; e += 256) {
        int n = e >> 8, dq = e & 255;
        float acc = 0.f;
        const float* mr = mem + n * 256;
        const float* wr = Wk + dq * 256;
        for (int i = 0; i < 256; ++i) acc += mr[i] * wr[i];
        k_bf[n * 256 + dq] = f2bf(acc);
    }

    // memT[d][n] = mem[n][d]
    for (int e = t; e < 256 * 64; e += 256) {
        int d = e >> 6, n = e & 63;
        memT_bf[e] = f2bf(mem[n * 256 + d]);
    }

    // bias[n] = max(log(importance * 0.99^age), -10)
    if (t < 64) {
        const float LOG_DECAY = -0.010050335853501441f; // log(0.99)
        float eff = imp[t] * expf(age[t] * LOG_DECAY);
        bias[t] = fmaxf(logf(eff), -10.0f);
    }
}

// ---------------------------------------------------------------------------
// Main fused kernel: 64 threads = 2 waves; each wave owns 32 rows (2 M-tiles).
// Phases: qW = query @ Wq^T (WMMA) -> LDS ; scores = qW @ k^T (WMMA) -> LDS ;
//         top-8 + softmax per row ; retrieved = attn @ mem (WMMA) -> global.
// ---------------------------------------------------------------------------
__global__ __launch_bounds__(64)
void mb_main(const float* __restrict__ query,
             const unsigned short* __restrict__ wq_bf,
             const unsigned short* __restrict__ k_bf,
             const unsigned short* __restrict__ memT_bf,
             const float* __restrict__ bias,
             float* __restrict__ out_ret,     // [ROWS][256]
             float* __restrict__ out_attn)    // [ROWS][64]
{
    __shared__ unsigned short sQW[64 * 256];    // qW bf16, 32 KB
    __shared__ float          sScores[64 * 64]; // 16 KB
    __shared__ unsigned short sAttn[64 * 64];   // 8 KB

    const int tid  = threadIdx.x;
    const int wave = tid >> 5;      // 0..1
    const int lane = tid & 31;
    const int m    = lane & 15;     // M (A) / N (B,C/D) lane index
    const int half = lane >> 4;     // lane half selects K group / M group
    const int wr0  = wave * 32;                     // wave's first row in WG
    const long long grb = (long long)blockIdx.x * 64; // WG's global row base

    // ---------------- Phase 1: projection qW = query @ Wq^T ----------------
    // A fragments (16x32 bf16 per K-tile): lane m holds row, half selects K.
    v16bf aq0[8], aq1[8];
    {
        const float* qrow0 = query + (grb + wr0 + m) * 256;
        const float* qrow1 = query + (grb + wr0 + 16 + m) * 256;
#pragma unroll
        for (int kt = 0; kt < 8; ++kt) {
            bfu fa0, fa1;
#pragma unroll
            for (int v = 0; v < 8; ++v) {
                int kk = kt * 32 + half * 16 + 2 * v;
                float2 p0 = *(const float2*)(qrow0 + kk);
                float2 p1 = *(const float2*)(qrow1 + kk);
                fa0.s[2 * v] = f2bf(p0.x); fa0.s[2 * v + 1] = f2bf(p0.y);
                fa1.s[2 * v] = f2bf(p1.x); fa1.s[2 * v + 1] = f2bf(p1.y);
            }
            aq0[kt] = fa0.v;
            aq1[kt] = fa1.v;
        }
    }

    const unsigned int* wq32 = (const unsigned int*)wq_bf;
#pragma unroll 4
    for (int nt = 0; nt < 16; ++nt) {
        v8f acc0 = {0.f, 0.f, 0.f, 0.f, 0.f, 0.f, 0.f, 0.f};
        v8f acc1 = {0.f, 0.f, 0.f, 0.f, 0.f, 0.f, 0.f, 0.f};
        const int dout = nt * 16 + m;               // B lane column
#pragma unroll
        for (int kt = 0; kt < 8; ++kt) {
            bfu fb;
            const int kb = kt * 32 + half * 16;     // B: lanes16-31 hold K+16
#pragma unroll
            for (int v = 0; v < 8; ++v)
                fb.u[v] = wq32[dout * 128 + (kb >> 1) + v]; // K-pair dword
            acc0 = WMMA_BF16(aq0[kt], fb.v, acc0);
            acc1 = WMMA_BF16(aq1[kt], fb.v, acc1);
        }
        // C/D layout: VGPR r -> row half*8+r, lane -> column
#pragma unroll
        for (int r = 0; r < 8; ++r) {
            sQW[(wr0 + half * 8 + r) * 256 + nt * 16 + m]      = f2bf(acc0[r]);
            sQW[(wr0 + 16 + half * 8 + r) * 256 + nt * 16 + m] = f2bf(acc1[r]);
        }
    }
    __syncthreads();

    // ---------------- Phase 2: scores = qW @ k^T / 16 ----------------------
    v16bf sa0[8], sa1[8];
    {
        const unsigned int* sq32 = (const unsigned int*)sQW;
#pragma unroll
        for (int kt = 0; kt < 8; ++kt) {
            bfu fa0, fa1;
#pragma unroll
            for (int v = 0; v < 8; ++v) {
                int kk = kt * 32 + half * 16 + 2 * v;
                fa0.u[v] = sq32[((wr0 + m) * 256 + kk) >> 1];
                fa1.u[v] = sq32[((wr0 + 16 + m) * 256 + kk) >> 1];
            }
            sa0[kt] = fa0.v;
            sa1[kt] = fa1.v;
        }
    }
    const unsigned int* k32 = (const unsigned int*)k_bf;
#pragma unroll
    for (int nt = 0; nt < 4; ++nt) {
        v8f acc0 = {0.f, 0.f, 0.f, 0.f, 0.f, 0.f, 0.f, 0.f};
        v8f acc1 = {0.f, 0.f, 0.f, 0.f, 0.f, 0.f, 0.f, 0.f};
        const int n = nt * 16 + m;
#pragma unroll
        for (int kt = 0; kt < 8; ++kt) {
            bfu fb;
            const int kb = kt * 32 + half * 16;
#pragma unroll
            for (int v = 0; v < 8; ++v)
                fb.u[v] = k32[n * 128 + (kb >> 1) + v];
            acc0 = WMMA_BF16(sa0[kt], fb.v, acc0);
            acc1 = WMMA_BF16(sa1[kt], fb.v, acc1);
        }
#pragma unroll
        for (int r = 0; r < 8; ++r) {
            sScores[(wr0 + half * 8 + r) * 64 + n]      = acc0[r] * 0.0625f;
            sScores[(wr0 + 16 + half * 8 + r) * 64 + n] = acc1[r] * 0.0625f;
        }
    }
    __syncthreads();

    // ---------------- Phase 3: top-8 mask + softmax (1 thread / row) -------
    {
        const int row = tid;                 // 64 threads <-> 64 rows
        unsigned long long sel = 0ull;
        float vals[8];
#pragma unroll 1
        for (int p = 0; p < 8; ++p) {
            float best = -3.4e38f;
            int bi = 0;
            for (int n = 0; n < 64; ++n) {
                if ((sel >> n) & 1ull) continue;
                float v = sScores[row * 64 + n] + bias[n];
                if (v > best) { best = v; bi = n; }   // ties -> lowest index
            }
            sel |= 1ull << bi;
            vals[p] = best;
        }
        const float mx = vals[0];
        float sum = 0.f;
#pragma unroll
        for (int p = 0; p < 8; ++p) sum += expf(vals[p] - mx);
        const float inv = 1.0f / sum;
        const long long grow = grb + row;
        for (int n = 0; n < 64; ++n) {
            float w = 0.f;
            if ((sel >> n) & 1ull)
                w = expf(sScores[row * 64 + n] + bias[n] - mx) * inv;
            out_attn[grow * 64 + n] = w;
            sAttn[row * 64 + n] = f2bf(w);
        }
    }
    __syncthreads();

    // ---------------- Phase 4: retrieved = attn @ mem ----------------------
    v16bf at0[2], at1[2];
    {
        const unsigned int* sa32 = (const unsigned int*)sAttn;
#pragma unroll
        for (int kt = 0; kt < 2; ++kt) {
            bfu fa0, fa1;
#pragma unroll
            for (int v = 0; v < 8; ++v) {
                int kk = kt * 32 + half * 16 + 2 * v;
                fa0.u[v] = sa32[((wr0 + m) * 64 + kk) >> 1];
                fa1.u[v] = sa32[((wr0 + 16 + m) * 64 + kk) >> 1];
            }
            at0[kt] = fa0.v;
            at1[kt] = fa1.v;
        }
    }
    const unsigned int* mt32 = (const unsigned int*)memT_bf;
#pragma unroll 4
    for (int nt = 0; nt < 16; ++nt) {
        v8f acc0 = {0.f, 0.f, 0.f, 0.f, 0.f, 0.f, 0.f, 0.f};
        v8f acc1 = {0.f, 0.f, 0.f, 0.f, 0.f, 0.f, 0.f, 0.f};
        const int d = nt * 16 + m;
#pragma unroll
        for (int kt = 0; kt < 2; ++kt) {
            bfu fb;
            const int kb = kt * 32 + half * 16;
#pragma unroll
            for (int v = 0; v < 8; ++v)
                fb.u[v] = mt32[d * 32 + (kb >> 1) + v];
            acc0 = WMMA_BF16(at0[kt], fb.v, acc0);
            acc1 = WMMA_BF16(at1[kt], fb.v, acc1);
        }
#pragma unroll
        for (int r = 0; r < 8; ++r) {
            out_ret[(grb + wr0 + half * 8 + r) * 256 + d]      = acc0[r];
            out_ret[(grb + wr0 + 16 + half * 8 + r) * 256 + d] = acc1[r];
        }
    }
}

// ---------------------------------------------------------------------------
extern "C" void kernel_launch(void* const* d_in, const int* in_sizes, int n_in,
                              void* d_out, int out_size, void* d_ws, size_t ws_size,
                              hipStream_t stream) {
    const float* query      = (const float*)d_in[0]; // [B,S,256]
    const float* memory     = (const float*)d_in[1]; // [1,64,256]
    const float* importance = (const float*)d_in[2]; // [1,64]
    const float* age        = (const float*)d_in[3]; // [1,64]
    const float* Wq         = (const float*)d_in[4]; // [256,256]
    const float* Wk         = (const float*)d_in[5]; // [256,256]
    // d_in[6] = top_k (fixed at 8 by the reference setup)

    const long long rows = (long long)in_sizes[0] / 256;   // B*S = 131072

    // Workspace layout (bytes): Wq bf16 | k bf16 | memT bf16 | bias f32
    unsigned short* wq_bf   = (unsigned short*)d_ws;
    unsigned short* k_bf    = wq_bf + 256 * 256;
    unsigned short* memT_bf = k_bf + 64 * 256;
    float*          bias    = (float*)(memT_bf + 256 * 64);

    float* out_ret  = (float*)d_out;
    float* out_attn = out_ret + rows * 256;

    mb_prep<<<1, 256, 0, stream>>>(memory, importance, age, Wq, Wk,
                                   wq_bf, k_bf, memT_bf, bias);

    const int nwg = (int)(rows / 64);
    mb_main<<<nwg, 64, 0, stream>>>(query, wq_bf, k_bf, memT_bf, bias,
                                    out_ret, out_attn);
}
